// SEGNNMessagePassing_2491081032269
// MI455X (gfx1250) — matrix operand order repaired and checked by reading, not compiled, not run on hardware
//
#include <hip/hip_runtime.h>

// ---------------------------------------------------------------------------
// SEGNN message passing for gfx1250 (MI455X).
// bf16 WMMA (v_wmma_f32_16x16x32_bf16) for all tensor-product GEMMs,
// f32 accumulate, LDS-staged A panels + transposed bf16 weights.
// ---------------------------------------------------------------------------

typedef __attribute__((ext_vector_type(16))) __bf16 v16bf;
typedef __attribute__((ext_vector_type(8)))  float  v8f;

#define INV_SQRT3 0.5773502691896258f
#define INV_SQRT2 0.7071067811865475f

union FragU { v16bf v; uint4 q[2]; };

__device__ __forceinline__ void wsync() {
  __builtin_amdgcn_wave_barrier();
  asm volatile("s_wait_dscnt 0" ::: "memory");
  __builtin_amdgcn_wave_barrier();
}

__device__ __forceinline__ float siluf(float x) { return x / (1.f + __expf(-x)); }
__device__ __forceinline__ float sigmf(float x) { return 1.f / (1.f + __expf(-x)); }

__device__ __forceinline__ void atomAddF(float* p, float v) {
  unsafeAtomicAdd(p, v);   // global_atomic_add_f32 (no return)
}

// A fragment: 16x32 bf16, row M = lane%16; lanes 0-15 hold K {k0..k0+7, k0+16..k0+23},
// lanes 16-31 hold K {+8}. Panel is [16][KPAD] bf16, K contiguous -> 2x b128 loads.
__device__ __forceinline__ v16bf ld_a(const __bf16* p) {
  FragU f;
  f.q[0] = *(const uint4*)(p);
  f.q[1] = *(const uint4*)(p + 16);
  return f.v;
}
// B fragment: 32x16 bf16, col N = lane%16; lanes 0-15 hold K {k0..k0+7, k0+16..k0+23},
// lanes 16-31 hold K {+8}. Weights stored transposed [N][KPAD] bf16.
__device__ __forceinline__ v16bf ld_b(const __bf16* p) {
  FragU f;
  f.q[0] = *(const uint4*)(p);
  f.q[1] = *(const uint4*)(p + 16);
  return f.v;
}

// C[j] += P[16 x KPAD] * W^T[(16*NT) x KPAD]   (accumulates into C)
template <int KPAD, int NT>
__device__ __forceinline__ void gemm_tile(v8f* C, const __bf16* P, const __bf16* W,
                                          int m16, int hi) {
#pragma unroll
  for (int k0 = 0; k0 < KPAD; k0 += 32) {
    v16bf a = ld_a(P + m16 * KPAD + k0 + 8 * hi);
#pragma unroll
    for (int j = 0; j < NT; ++j) {
      v16bf b = ld_b(W + (j * 16 + m16) * KPAD + k0 + 8 * hi);
      C[j] = __builtin_amdgcn_wmma_f32_16x16x32_bf16(false, a, false, b,
                                                     (short)0, C[j], false, false);
    }
  }
}

// ---------------------------------------------------------------------------
__global__ void zero_f4(float4* p, long n) {
  long i = (long)blockIdx.x * blockDim.x + threadIdx.x;
  long stride = (long)gridDim.x * blockDim.x;
  for (; i < n; i += stride) p[i] = make_float4(0.f, 0.f, 0.f, 0.f);
}

// ---------------------------------------------------------------------------
// Edge kernel: two gated tensor products + atomic scatter-add into agg buffers.
// ---------------------------------------------------------------------------
struct EMeta {
  int   dst[16];
  int   src[16];
  float ys[16];
  float valid[16];
  float yv[16][3];
};

#define EW 2   // waves per block (LDS budget: 49KB weights + 2*5.5KB staging)

__global__ __launch_bounds__(EW * 32) void segnn_edge(
    const float* __restrict__ node_s, const float* __restrict__ node_v,
    const float* __restrict__ edge_emb, const float* __restrict__ eas,
    const float* __restrict__ eav, const int* __restrict__ eidx,
    long E,
    const float* __restrict__ w1_ss, const float* __restrict__ w1_vv0,
    const float* __restrict__ w1_sv, const float* __restrict__ w1_vs,
    const float* __restrict__ w1_vv1,
    const float* __restrict__ w2_ss, const float* __restrict__ w2_vv0,
    const float* __restrict__ w2_sv, const float* __restrict__ w2_vs,
    const float* __restrict__ w2_vv1,
    float* __restrict__ aggS, float* __restrict__ aggV) {
  // Transposed bf16 weights in LDS: [n_out][k_pad]
  __shared__ __attribute__((aligned(16))) __bf16 sW1A[64 * 160];  // [w1_ss;w1_vv0]
  __shared__ __attribute__((aligned(16))) __bf16 sW1B[32 * 96];   // w1_sv
  __shared__ __attribute__((aligned(16))) __bf16 sW1C[32 * 128];  // [w1_vs;w1_vv1]
  __shared__ __attribute__((aligned(16))) __bf16 sW2A[64 * 64];   // [w2_ss;w2_vv0]
  __shared__ __attribute__((aligned(16))) __bf16 sW2B[32 * 32];   // w2_sv
  __shared__ __attribute__((aligned(16))) __bf16 sW2C[32 * 64];   // [w2_vs;w2_vv1]
  __shared__ __attribute__((aligned(16))) __bf16 sP[EW][16 * 160];
  __shared__ EMeta sM[EW];

  const int tid = threadIdx.x, nth = blockDim.x;
  for (int i = tid; i < 64 * 160; i += nth) {
    int n = i / 160, k = i - n * 160;
    float v = (k < 80) ? w1_ss[k * 64 + n] : (k < 144 ? w1_vv0[(k - 80) * 64 + n] : 0.f);
    sW1A[i] = (__bf16)v;
  }
  for (int i = tid; i < 32 * 96; i += nth) {
    int n = i / 96, k = i - n * 96;
    sW1B[i] = (__bf16)((k < 80) ? w1_sv[k * 32 + n] : 0.f);
  }
  for (int i = tid; i < 32 * 128; i += nth) {
    int n = i / 128, k = i - n * 128;
    sW1C[i] = (__bf16)((k < 64) ? w1_vs[k * 32 + n] : w1_vv1[(k - 64) * 32 + n]);
  }
  for (int i = tid; i < 64 * 64; i += nth) {
    int n = i / 64, k = i - n * 64;
    sW2A[i] = (__bf16)((k < 32) ? w2_ss[k * 64 + n] : w2_vv0[(k - 32) * 64 + n]);
  }
  for (int i = tid; i < 32 * 32; i += nth) {
    int n = i / 32, k = i - n * 32;
    sW2B[i] = (__bf16)w2_sv[k * 32 + n];
  }
  for (int i = tid; i < 32 * 64; i += nth) {
    int n = i / 64, k = i - n * 64;
    sW2C[i] = (__bf16)((k < 32) ? w2_vs[k * 32 + n] : w2_vv1[(k - 32) * 32 + n]);
  }
  __syncthreads();

  const int wave = tid >> 5, lane = tid & 31;
  const int m16 = lane & 15, hi = lane >> 4;
  __bf16* P = sP[wave];
  EMeta& M = sM[wave];

  const float INV_S1 = 1.f / 12.f;             // 1/sqrt(80+64)
  const float INV_V1 = 0.06933752452815365f;   // 1/sqrt(80+128)
  const float INV_S2 = 0.125f;                 // 1/sqrt(32+32)
  const float INV_V2 = 0.10206207261596575f;   // 1/sqrt(32+64)

  long numTiles = (E + 15) >> 4;
  for (long tile = (long)blockIdx.x * EW + wave; tile < numTiles;
       tile += (long)gridDim.x * EW) {
    // ---- tile metadata ----
    if (lane < 16) {
      long e = tile * 16 + lane;
      bool ok = e < E;
      long ec = ok ? e : (E - 1);
      M.src[lane]   = ok ? eidx[ec] : 0;
      M.dst[lane]   = ok ? eidx[E + ec] : 0;
      M.ys[lane]    = ok ? eas[ec] : 0.f;
      M.valid[lane] = ok ? 1.f : 0.f;
      M.yv[lane][0] = ok ? eav[ec * 3 + 0] : 0.f;
      M.yv[lane][1] = ok ? eav[ec * 3 + 1] : 0.f;
      M.yv[lane][2] = ok ? eav[ec * 3 + 2] : 0.f;
    }
    wsync();

    // ---- G1 panel: [xs*ys (K=80) | dot/sqrt3 (K=64)], Kpad=160 ----
    for (int i = lane; i < 16 * 160; i += 32) {
      int m = i / 160, k = i - m * 160;
      float val = 0.f;
      if (k < 80) {
        float x;
        if (k < 32)      x = node_s[(long)M.dst[m] * 32 + k];
        else if (k < 64) x = node_s[(long)M.src[m] * 32 + (k - 32)];
        else {
          long e = tile * 16 + m; if (e >= E) e = E - 1;
          x = edge_emb[e * 16 + (k - 64)];
        }
        val = x * M.ys[m];
      } else if (k < 144) {
        int j = k - 80;
        const float* xv = (j < 32) ? node_v + (long)M.dst[m] * 96 + j * 3
                                   : node_v + (long)M.src[m] * 96 + (j - 32) * 3;
        val = (xv[0] * M.yv[m][0] + xv[1] * M.yv[m][1] + xv[2] * M.yv[m][2]) * INV_SQRT3;
      }
      P[i] = (__bf16)val;
    }
    wsync();
    v8f os[4] = {};
    gemm_tile<160, 4>(os, P, sW1A, m16, hi);

    float msf[2][8], gatef[2][8];
#pragma unroll
    for (int j = 0; j < 2; ++j)
#pragma unroll
      for (int g = 0; g < 8; ++g) {
        msf[j][g]   = siluf(os[j][g] * INV_S1);
        gatef[j][g] = sigmf(os[j + 2][g] * INV_S1);
      }

    // ---- G2 panel: xs (K=80), Kpad=96 ----
    wsync();
    for (int i = lane; i < 16 * 96; i += 32) {
      int m = i / 96, k = i - m * 96;
      float val = 0.f;
      if (k < 80) {
        float x;
        if (k < 32)      x = node_s[(long)M.dst[m] * 32 + k];
        else if (k < 64) x = node_s[(long)M.src[m] * 32 + (k - 32)];
        else {
          long e = tile * 16 + m; if (e >= E) e = E - 1;
          x = edge_emb[e * 16 + (k - 64)];
        }
        val = x * M.valid[m];
      }
      P[i] = (__bf16)val;
    }
    wsync();
    v8f tsv[2] = {};
    gemm_tile<96, 2>(tsv, P, sW1B, m16, hi);

    // ---- G3 per component: [xv_c*ys | cross_c/sqrt2], Kpad=128 ----
    float mvf[3][2][8];
#pragma unroll 1
    for (int c = 0; c < 3; ++c) {
      int c1 = (c + 1) % 3, c2 = (c + 2) % 3;
      wsync();
      for (int i = lane; i < 16 * 128; i += 32) {
        int m = i / 128, k = i - m * 128;
        float val;
        if (k < 64) {
          const float* xv = (k < 32) ? node_v + (long)M.dst[m] * 96 + k * 3
                                     : node_v + (long)M.src[m] * 96 + (k - 32) * 3;
          val = xv[c] * M.ys[m];
        } else {
          int j = k - 64;
          const float* xv = (j < 32) ? node_v + (long)M.dst[m] * 96 + j * 3
                                     : node_v + (long)M.src[m] * 96 + (j - 32) * 3;
          val = (xv[c1] * M.yv[m][c2] - xv[c2] * M.yv[m][c1]) * INV_SQRT2;
        }
        P[i] = (__bf16)val;
      }
      wsync();
      v8f t3[2] = {};
      gemm_tile<128, 2>(t3, P, sW1C, m16, hi);
#pragma unroll
      for (int j = 0; j < 2; ++j)
#pragma unroll
        for (int g = 0; g < 8; ++g) {
          int Mr = g + 8 * hi;
          mvf[c][j][g] = (tsv[j][g] * M.yv[Mr][c] + t3[j][g]) * INV_V1 * gatef[j][g];
        }
    }

    // ================= layer 2 =================
    // H1 panel: [ms*ys | dot2/sqrt3], Kpad=64 (built from register fragments)
    wsync();
#pragma unroll
    for (int j = 0; j < 2; ++j)
#pragma unroll
      for (int g = 0; g < 8; ++g) {
        int Mr = g + 8 * hi, col = j * 16 + m16;
        P[Mr * 64 + col] = (__bf16)(msf[j][g] * M.ys[Mr]);
        float d2 = (mvf[0][j][g] * M.yv[Mr][0] + mvf[1][j][g] * M.yv[Mr][1] +
                    mvf[2][j][g] * M.yv[Mr][2]) * INV_SQRT3;
        P[Mr * 64 + 32 + col] = (__bf16)d2;
      }
    wsync();
    v8f os2[4] = {};
    gemm_tile<64, 4>(os2, P, sW2A, m16, hi);

    float ms2f[2][8], gate2f[2][8];
#pragma unroll
    for (int j = 0; j < 2; ++j)
#pragma unroll
      for (int g = 0; g < 8; ++g) {
        ms2f[j][g]   = siluf(os2[j][g] * INV_S2);
        gate2f[j][g] = sigmf(os2[j + 2][g] * INV_S2);
      }

    // H2 panel: ms, Kpad=32
    wsync();
#pragma unroll
    for (int j = 0; j < 2; ++j)
#pragma unroll
      for (int g = 0; g < 8; ++g) {
        int Mr = g + 8 * hi;
        P[Mr * 32 + j * 16 + m16] = (__bf16)msf[j][g];
      }
    wsync();
    v8f tsv2[2] = {};
    gemm_tile<32, 2>(tsv2, P, sW2B, m16, hi);

    // scatter scalar messages
#pragma unroll
    for (int j = 0; j < 2; ++j)
#pragma unroll
      for (int g = 0; g < 8; ++g) {
        int Mr = g + 8 * hi;
        if (M.valid[Mr] > 0.f)
          atomAddF(&aggS[(long)M.dst[Mr] * 32 + j * 16 + m16], ms2f[j][g]);
      }

    // H3 per component + scatter vector messages
#pragma unroll 1
    for (int c = 0; c < 3; ++c) {
      int c1 = (c + 1) % 3, c2 = (c + 2) % 3;
      wsync();
#pragma unroll
      for (int j = 0; j < 2; ++j)
#pragma unroll
        for (int g = 0; g < 8; ++g) {
          int Mr = g + 8 * hi, col = j * 16 + m16;
          P[Mr * 64 + col] = (__bf16)(mvf[c][j][g] * M.ys[Mr]);
          float cr = (mvf[c1][j][g] * M.yv[Mr][c2] - mvf[c2][j][g] * M.yv[Mr][c1]) * INV_SQRT2;
          P[Mr * 64 + 32 + col] = (__bf16)cr;
        }
      wsync();
      v8f t3[2] = {};
      gemm_tile<64, 2>(t3, P, sW2C, m16, hi);
#pragma unroll
      for (int j = 0; j < 2; ++j)
#pragma unroll
        for (int g = 0; g < 8; ++g) {
          int Mr = g + 8 * hi;
          float v = (tsv2[j][g] * M.yv[Mr][c] + t3[j][g]) * INV_V2 * gate2f[j][g];
          if (M.valid[Mr] > 0.f)
            atomAddF(&aggV[(long)M.dst[Mr] * 96 + (j * 16 + m16) * 3 + c], v);
        }
    }
    wsync();
  }
}

// ---------------------------------------------------------------------------
// Node kernel: gated TP (u1) + plain FCTP (u2) + residual, writes d_out[N,128].
// ---------------------------------------------------------------------------
struct NMeta {
  int   idx[16];
  float ys[16];
  float valid[16];
  float yv[16][3];
};

#define NW 4

__global__ __launch_bounds__(NW * 32) void segnn_node(
    const float* __restrict__ node_s, const float* __restrict__ node_v,
    const float* __restrict__ nas, const float* __restrict__ nav,
    const float* __restrict__ aggS, const float* __restrict__ aggV, long N,
    const float* __restrict__ u1_ss, const float* __restrict__ u1_vv0,
    const float* __restrict__ u1_sv, const float* __restrict__ u1_vs,
    const float* __restrict__ u1_vv1,
    const float* __restrict__ u2_ss, const float* __restrict__ u2_vv0,
    const float* __restrict__ u2_sv, const float* __restrict__ u2_vs,
    const float* __restrict__ u2_vv1,
    float* __restrict__ out) {
  __shared__ __attribute__((aligned(16))) __bf16 sUA[64 * 128];  // [u1_ss;u1_vv0]
  __shared__ __attribute__((aligned(16))) __bf16 sUB[32 * 64];   // u1_sv
  __shared__ __attribute__((aligned(16))) __bf16 sUC[32 * 128];  // [u1_vs;u1_vv1]
  __shared__ __attribute__((aligned(16))) __bf16 sVA[32 * 64];   // [u2_ss;u2_vv0]
  __shared__ __attribute__((aligned(16))) __bf16 sVB[32 * 32];   // u2_sv
  __shared__ __attribute__((aligned(16))) __bf16 sVC[32 * 64];   // [u2_vs;u2_vv1]
  __shared__ __attribute__((aligned(16))) __bf16 sP[NW][16 * 128];
  __shared__ NMeta sM[NW];

  const int tid = threadIdx.x, nth = blockDim.x;
  for (int i = tid; i < 64 * 128; i += nth) {
    int n = i / 128, k = i - n * 128;
    sUA[i] = (__bf16)((k < 64) ? u1_ss[k * 64 + n] : u1_vv0[(k - 64) * 64 + n]);
  }
  for (int i = tid; i < 32 * 64; i += nth) {
    int n = i / 64, k = i - n * 64;
    sUB[i] = (__bf16)u1_sv[k * 32 + n];
  }
  for (int i = tid; i < 32 * 128; i += nth) {
    int n = i / 128, k = i - n * 128;
    sUC[i] = (__bf16)((k < 64) ? u1_vs[k * 32 + n] : u1_vv1[(k - 64) * 32 + n]);
  }
  for (int i = tid; i < 32 * 64; i += nth) {
    int n = i / 64, k = i - n * 64;
    sVA[i] = (__bf16)((k < 32) ? u2_ss[k * 32 + n] : u2_vv0[(k - 32) * 32 + n]);
  }
  for (int i = tid; i < 32 * 32; i += nth) {
    int n = i / 32, k = i - n * 32;
    sVB[i] = (__bf16)u2_sv[k * 32 + n];
  }
  for (int i = tid; i < 32 * 64; i += nth) {
    int n = i / 64, k = i - n * 64;
    sVC[i] = (__bf16)((k < 32) ? u2_vs[k * 32 + n] : u2_vv1[(k - 32) * 32 + n]);
  }
  __syncthreads();

  const int wave = tid >> 5, lane = tid & 31;
  const int m16 = lane & 15, hi = lane >> 4;
  __bf16* P = sP[wave];
  NMeta& M = sM[wave];

  const float INV_S1 = 0.08838834764831845f;  // 1/sqrt(64+64)
  const float INV_V1 = 0.07216878364870323f;  // 1/sqrt(64+128)
  const float INV_S2 = 0.125f;                // 1/sqrt(64)
  const float INV_V2 = 0.10206207261596575f;  // 1/sqrt(96)

  long numTiles = (N + 15) >> 4;
  for (long tile = (long)blockIdx.x * NW + wave; tile < numTiles;
       tile += (long)gridDim.x * NW) {
    if (lane < 16) {
      long n = tile * 16 + lane;
      bool ok = n < N;
      long nc = ok ? n : (N - 1);
      M.idx[lane]   = (int)nc;
      M.ys[lane]    = ok ? nas[nc] : 0.f;
      M.valid[lane] = ok ? 1.f : 0.f;
      M.yv[lane][0] = ok ? nav[nc * 3 + 0] : 0.f;
      M.yv[lane][1] = ok ? nav[nc * 3 + 1] : 0.f;
      M.yv[lane][2] = ok ? nav[nc * 3 + 2] : 0.f;
    }
    wsync();

    // G1: [us*ys (K=64) | dot/sqrt3 (K=64)], Kpad=128
    for (int i = lane; i < 16 * 128; i += 32) {
      int m = i / 128, k = i - m * 128;
      long nn = M.idx[m];
      float val;
      if (k < 64) {
        float x = (k < 32) ? node_s[nn * 32 + k] : aggS[nn * 32 + (k - 32)];
        val = x * M.ys[m];
      } else {
        int j = k - 64;
        const float* xv = (j < 32) ? node_v + nn * 96 + j * 3
                                   : aggV + nn * 96 + (j - 32) * 3;
        val = (xv[0] * M.yv[m][0] + xv[1] * M.yv[m][1] + xv[2] * M.yv[m][2]) * INV_SQRT3;
      }
      P[i] = (__bf16)val;
    }
    wsync();
    v8f os[4] = {};
    gemm_tile<128, 4>(os, P, sUA, m16, hi);

    float msf[2][8], gatef[2][8];
#pragma unroll
    for (int j = 0; j < 2; ++j)
#pragma unroll
      for (int g = 0; g < 8; ++g) {
        msf[j][g]   = siluf(os[j][g] * INV_S1);
        gatef[j][g] = sigmf(os[j + 2][g] * INV_S1);
      }

    // G2: us, Kpad=64
    wsync();
    for (int i = lane; i < 16 * 64; i += 32) {
      int m = i / 64, k = i - m * 64;
      long nn = M.idx[m];
      float x = (k < 32) ? node_s[nn * 32 + k] : aggS[nn * 32 + (k - 32)];
      P[i] = (__bf16)(x * M.valid[m]);
    }
    wsync();
    v8f tsv[2] = {};
    gemm_tile<64, 2>(tsv, P, sUB, m16, hi);

    // G3 per component
    float mvf[3][2][8];
#pragma unroll 1
    for (int c = 0; c < 3; ++c) {
      int c1 = (c + 1) % 3, c2 = (c + 2) % 3;
      wsync();
      for (int i = lane; i < 16 * 128; i += 32) {
        int m = i / 128, k = i - m * 128;
        long nn = M.idx[m];
        float val;
        if (k < 64) {
          const float* xv = (k < 32) ? node_v + nn * 96 + k * 3
                                     : aggV + nn * 96 + (k - 32) * 3;
          val = xv[c] * M.ys[m];
        } else {
          int j = k - 64;
          const float* xv = (j < 32) ? node_v + nn * 96 + j * 3
                                     : aggV + nn * 96 + (j - 32) * 3;
          val = (xv[c1] * M.yv[m][c2] - xv[c2] * M.yv[m][c1]) * INV_SQRT2;
        }
        P[i] = (__bf16)val;
      }
      wsync();
      v8f t3[2] = {};
      gemm_tile<128, 2>(t3, P, sUC, m16, hi);
#pragma unroll
      for (int j = 0; j < 2; ++j)
#pragma unroll
        for (int g = 0; g < 8; ++g) {
          int Mr = g + 8 * hi;
          mvf[c][j][g] = (tsv[j][g] * M.yv[Mr][c] + t3[j][g]) * INV_V1 * gatef[j][g];
        }
    }

    // ===== u2 (plain FCTP) =====
    wsync();
#pragma unroll
    for (int j = 0; j < 2; ++j)
#pragma unroll
      for (int g = 0; g < 8; ++g) {
        int Mr = g + 8 * hi, col = j * 16 + m16;
        P[Mr * 64 + col] = (__bf16)(msf[j][g] * M.ys[Mr]);
        float d2 = (mvf[0][j][g] * M.yv[Mr][0] + mvf[1][j][g] * M.yv[Mr][1] +
                    mvf[2][j][g] * M.yv[Mr][2]) * INV_SQRT3;
        P[Mr * 64 + 32 + col] = (__bf16)d2;
      }
    wsync();
    v8f osf[2] = {};
    gemm_tile<64, 2>(osf, P, sVA, m16, hi);

    wsync();
#pragma unroll
    for (int j = 0; j < 2; ++j)
#pragma unroll
      for (int g = 0; g < 8; ++g) {
        int Mr = g + 8 * hi;
        P[Mr * 32 + j * 16 + m16] = (__bf16)msf[j][g];
      }
    wsync();
    v8f tsv2[2] = {};
    gemm_tile<32, 2>(tsv2, P, sVB, m16, hi);

    // scalar output + residual
#pragma unroll
    for (int j = 0; j < 2; ++j)
#pragma unroll
      for (int g = 0; g < 8; ++g) {
        int Mr = g + 8 * hi;
        if (M.valid[Mr] > 0.f) {
          long nn = M.idx[Mr];
          int col = j * 16 + m16;
          out[nn * 128 + col] = osf[j][g] * INV_S2 + node_s[nn * 32 + col];
        }
      }

    // vector output per component + residual
#pragma unroll 1
    for (int c = 0; c < 3; ++c) {
      int c1 = (c + 1) % 3, c2 = (c + 2) % 3;
      wsync();
#pragma unroll
      for (int j = 0; j < 2; ++j)
#pragma unroll
        for (int g = 0; g < 8; ++g) {
          int Mr = g + 8 * hi, col = j * 16 + m16;
          P[Mr * 64 + col] = (__bf16)(mvf[c][j][g] * M.ys[Mr]);
          float cr = (mvf[c1][j][g] * M.yv[Mr][c2] - mvf[c2][j][g] * M.yv[Mr][c1]) * INV_SQRT2;
          P[Mr * 64 + 32 + col] = (__bf16)cr;
        }
      wsync();
      v8f t3[2] = {};
      gemm_tile<64, 2>(t3, P, sVC, m16, hi);
#pragma unroll
      for (int j = 0; j < 2; ++j)
#pragma unroll
        for (int g = 0; g < 8; ++g) {
          int Mr = g + 8 * hi;
          if (M.valid[Mr] > 0.f) {
            long nn = M.idx[Mr];
            int col = j * 16 + m16;
            float v = (tsv2[j][g] * M.yv[Mr][c] + t3[j][g]) * INV_V2;
            out[nn * 128 + 32 + col * 3 + c] = v + node_v[nn * 96 + col * 3 + c];
          }
        }
    }
    wsync();
  }
}

// ---------------------------------------------------------------------------
extern "C" void kernel_launch(void* const* d_in, const int* in_sizes, int n_in,
                              void* d_out, int out_size, void* d_ws, size_t ws_size,
                              hipStream_t stream) {
  const float* node_s   = (const float*)d_in[0];
  const float* node_v   = (const float*)d_in[1];
  const float* nas      = (const float*)d_in[2];
  const float* nav      = (const float*)d_in[3];
  const float* edge_emb = (const float*)d_in[4];
  const float* eas      = (const float*)d_in[5];
  const float* eav      = (const float*)d_in[6];
  const int*   eidx     = (const int*)d_in[7];
  // d_in[8] = batch (unused)
  const float* W[20];
  for (int i = 0; i < 20; ++i) W[i] = (const float*)d_in[9 + i];

  long N = in_sizes[2];
  long E = in_sizes[5];

  float* aggS = (float*)d_ws;                 // [N,32]
  float* aggV = aggS + (size_t)N * 32;        // [N,32,3]

  // zero accumulators (d_ws is poisoned by the harness)
  long tot4 = N * 32;  // float4 count for N*128 floats
  long zg = (tot4 + 255) / 256; if (zg > 8192) zg = 8192;
  zero_f4<<<dim3((unsigned)zg), dim3(256), 0, stream>>>((float4*)d_ws, tot4);

  long eTiles = (E + 15) / 16;
  long eg = (eTiles + EW - 1) / EW; if (eg > 2048) eg = 2048;
  segnn_edge<<<dim3((unsigned)eg), dim3(EW * 32), 0, stream>>>(
      node_s, node_v, edge_emb, eas, eav, eidx, E,
      W[0], W[1], W[2], W[3], W[4], W[5], W[6], W[7], W[8], W[9],
      aggS, aggV);

  long nTiles = (N + 15) / 16;
  long ng = (nTiles + NW - 1) / NW; if (ng > 2048) ng = 2048;
  segnn_node<<<dim3((unsigned)ng), dim3(NW * 32), 0, stream>>>(
      node_s, node_v, nas, nav, aggS, aggV, N,
      W[10], W[11], W[12], W[13], W[14], W[15], W[16], W[17], W[18], W[19],
      (float*)d_out);
}
